// BertSelfAttention_1357209666213
// MI455X (gfx1250) — matrix operand order repaired and checked by reading, not compiled, not run on hardware
//
#include <hip/hip_runtime.h>

typedef __attribute__((ext_vector_type(8)))  _Float16 v8h;
typedef __attribute__((ext_vector_type(16))) _Float16 v16h;
typedef __attribute__((ext_vector_type(8)))  float    v8f;

#define B_  4
#define S_  2048
#define H_  1024
#define NH  16
#define HD  64

// ---- CDNA5 async global->LDS copy (ASYNCcnt), GVS addressing ------------
__device__ __forceinline__ void async_copy_b128(unsigned lds_off, unsigned voff,
                                                unsigned long long sbase) {
    asm volatile("global_load_async_to_lds_b128 %0, %1, %2"
                 :: "v"(lds_off), "v"(voff), "s"(sbase) : "memory");
}
__device__ __forceinline__ void wait_async0() {
    asm volatile("s_wait_asynccnt 0x0" ::: "memory");
}
__device__ __forceinline__ unsigned lds_off(const void* p) {
    return (unsigned)(unsigned long long)p;   // low 32 bits = LDS byte offset
}

// ---- WMMA helpers (gfx1250, wave32) -------------------------------------
__device__ __forceinline__ v8f wmma_f16(v16h a, v16h b, v8f c) {
    return __builtin_amdgcn_wmma_f32_16x16x32_f16(
        false, a, false, b, (short)0, c, false, false);
}

// A fragment: lane holds row m = lane%16; halves: k = g*8+e (e<8), 16+g*8+e-8
__device__ __forceinline__ v16h load_afrag(const _Float16* row, int g) {
    v8h lo = *(const v8h*)(row + g * 8);
    v8h hi = *(const v8h*)(row + 16 + g * 8);
    v16h a;
#pragma unroll
    for (int i = 0; i < 8; ++i) { a[i] = lo[i]; a[i + 8] = hi[i]; }
    return a;
}

// B fragment from [n][k] layout: lane holds col n = lane%16; halves: k = g*8+e (e<8), 16+g*8+e-8
__device__ __forceinline__ v16h load_bfrag(const _Float16* row, int g) {
    v8h lo = *(const v8h*)(row + g * 8);
    v8h hi = *(const v8h*)(row + 16 + g * 8);
    v16h b;
#pragma unroll
    for (int i = 0; i < 8; ++i) { b[i] = lo[i]; b[i + 8] = hi[i]; }
    return b;
}

// ---- Kernel 1: f32 -> f16 convert (hidden states) -----------------------
__global__ void cvt_f32_f16(const float* __restrict__ x, _Float16* __restrict__ y, int n8) {
    int i = blockIdx.x * blockDim.x + threadIdx.x;
    if (i >= n8) return;
    const float4* p = (const float4*)x + (size_t)i * 2;
    float4 a = p[0], c = p[1];
    v8h o;
    o[0] = (_Float16)a.x; o[1] = (_Float16)a.y; o[2] = (_Float16)a.z; o[3] = (_Float16)a.w;
    o[4] = (_Float16)c.x; o[5] = (_Float16)c.y; o[6] = (_Float16)c.z; o[7] = (_Float16)c.w;
    ((v8h*)y)[i] = o;
}

// ---- Kernel 2: W[k][n] f32 -> Wt[n][k] f16 (tiled transpose) ------------
__global__ void transpose_w(const float* __restrict__ Wq, const float* __restrict__ Wk,
                            const float* __restrict__ Wv,
                            _Float16* __restrict__ Wtq, _Float16* __restrict__ Wtk,
                            _Float16* __restrict__ Wtv) {
    const float* W = blockIdx.z == 0 ? Wq : (blockIdx.z == 1 ? Wk : Wv);
    _Float16* Wt   = blockIdx.z == 0 ? Wtq : (blockIdx.z == 1 ? Wtk : Wtv);
    __shared__ float t[32][33];
    int k0 = blockIdx.y * 32, n0 = blockIdx.x * 32;
    int tx = threadIdx.x, ty = threadIdx.y;      // 32 x 8
#pragma unroll
    for (int i = 0; i < 4; ++i)
        t[ty + 8 * i][tx] = W[(size_t)(k0 + ty + 8 * i) * H_ + n0 + tx];
    __syncthreads();
#pragma unroll
    for (int i = 0; i < 4; ++i)
        Wt[(size_t)(n0 + ty + 8 * i) * H_ + k0 + tx] = (_Float16)t[tx][ty + 8 * i];
}

// ---- Kernel 3: fused QKV GEMM, K=64 stages, async double buffering ------
__global__ void __launch_bounds__(256)
qkv_gemm(const _Float16* __restrict__ Xh,
         const _Float16* __restrict__ Wtq, const _Float16* __restrict__ Wtk,
         const _Float16* __restrict__ Wtv,
         const float* __restrict__ bq, const float* __restrict__ bk,
         const float* __restrict__ bv,
         _Float16* __restrict__ Qo, _Float16* __restrict__ Ko, _Float16* __restrict__ Vto)
{
    const int which = blockIdx.z;
    const _Float16* Wt = which == 0 ? Wtq : (which == 1 ? Wtk : Wtv);
    const float* bias  = which == 0 ? bq  : (which == 1 ? bk  : bv);
    _Float16* Out      = which == 0 ? Qo  : (which == 1 ? Ko  : Vto);

    __shared__ __align__(16) _Float16 As[2][128][72];   // [buf][m][k0..63]
    __shared__ __align__(16) _Float16 Bs[2][128][72];   // [buf][n][k0..63]

    const int tid  = threadIdx.x;
    const int lane = tid & 31;
    const int wave = tid >> 5;
    const int n16  = lane & 15;
    const int g    = lane >> 4;
    const int wm   = wave >> 1;         // m offset wm*32
    const int wn   = wave & 1;          // n offset wn*64
    const int mtile = blockIdx.y;
    const int ntile = blockIdx.x;

    const unsigned long long xb = (unsigned long long)Xh;
    const unsigned long long wb = (unsigned long long)Wt;

    // 1024 16B-chunks per 128x64 tile; 4 chunks per thread per matrix
    const int ccol = (tid & 7) * 8;     // half offset within the 64-wide k slab

    v8f acc[2][4];
#pragma unroll
    for (int i = 0; i < 2; ++i)
#pragma unroll
        for (int j = 0; j < 4; ++j) acc[i][j] = (v8f)(0.0f);

    auto issue = [&](int k0, int buf) {
#pragma unroll
        for (int q = 0; q < 4; ++q) {
            int row = (tid + q * 256) >> 3;
            async_copy_b128(lds_off(&As[buf][row][ccol]),
                            (unsigned)(((mtile * 128 + row) * H_ + k0 + ccol) * 2), xb);
            async_copy_b128(lds_off(&Bs[buf][row][ccol]),
                            (unsigned)(((ntile * 128 + row) * H_ + k0 + ccol) * 2), wb);
        }
    };

    issue(0, 0);
    for (int k0 = 0, buf = 0; k0 < H_; k0 += 64, buf ^= 1) {
        wait_async0();          // my tile chunks have landed in LDS
        __syncthreads();        // everyone's chunks have landed
        if (k0 + 64 < H_) issue(k0 + 64, buf ^ 1);   // overlap with math

#pragma unroll
        for (int ks = 0; ks < 64; ks += 32) {
            v16h a[2], bf[4];
#pragma unroll
            for (int mf = 0; mf < 2; ++mf)
                a[mf] = load_afrag(&As[buf][wm * 32 + mf * 16 + n16][ks], g);
#pragma unroll
            for (int nf = 0; nf < 4; ++nf)
                bf[nf] = load_bfrag(&Bs[buf][wn * 64 + nf * 16 + n16][ks], g);
#pragma unroll
            for (int mf = 0; mf < 2; ++mf)
#pragma unroll
                for (int nf = 0; nf < 4; ++nf)
                    acc[mf][nf] = wmma_f16(a[mf], bf[nf], acc[mf][nf]);
        }
        __syncthreads();        // done reading buf before it is refilled
    }

    // epilogue: bias add, cast f16, scatter to attention-friendly layouts
#pragma unroll
    for (int nf = 0; nf < 4; ++nf) {
        int nG = ntile * 128 + wn * 64 + nf * 16 + n16;   // N = h*64+hd
        float bsc = bias[nG];
        int h = nG >> 6, hd = nG & 63;
#pragma unroll
        for (int mf = 0; mf < 2; ++mf) {
            int mbase = mtile * 128 + wm * 32 + mf * 16 + 8 * g;
            int bb = mbase >> 11;
            int sbase = mbase & 2047;
            if (which == 2) {
                v8h pk;
#pragma unroll
                for (int r = 0; r < 8; ++r) pk[r] = (_Float16)(acc[mf][nf][r] + bsc);
                *(v8h*)(Out + ((size_t)(bb * NH + h) * HD + hd) * S_ + sbase) = pk;
            } else {
#pragma unroll
                for (int r = 0; r < 8; ++r)
                    Out[((size_t)(bb * NH + h) * S_ + (sbase + r)) * HD + hd] =
                        (_Float16)(acc[mf][nf][r] + bsc);
            }
        }
    }
}

// ---- Kernel 4: flash attention, async double-buffered K/V tiles ---------
__global__ void __launch_bounds__(128)
flash_attn(const _Float16* __restrict__ Q, const _Float16* __restrict__ K,
           const _Float16* __restrict__ Vt, const float* __restrict__ mask,
           float* __restrict__ out)
{
    __shared__ __align__(16) _Float16 Ks[2][32][72];     // [buf][key][hd]
    __shared__ __align__(16) _Float16 Vs[2][64][40];     // [buf][hd][key]
    __shared__ __align__(16) _Float16 Pl[4][16][40];     // per-wave P tile

    const int tid  = threadIdx.x;       // 128 = 4 waves
    const int lane = tid & 31;
    const int w    = tid >> 5;
    const int n16  = lane & 15;
    const int g    = lane >> 4;
    const int qt = blockIdx.x, h = blockIdx.y, b = blockIdx.z;
    const int qrow0 = qt * 64 + w * 16;

    const _Float16* Qg = Q  + (size_t)(b * NH + h) * S_ * HD;
    const unsigned long long kb64 = (unsigned long long)(K  + (size_t)(b * NH + h) * S_ * HD);
    const unsigned long long vb64 = (unsigned long long)(Vt + (size_t)(b * NH + h) * HD * S_);
    const float* mb = mask + (size_t)b * S_;

    // async staging coords: K tile 32x64 (8 x 16B chunks/row), V tile 64x32
    const int kr = tid >> 2, kc = (tid & 3) * 16;
    const int vr = tid >> 1, vc = (tid & 1) * 16;
    auto issue_kv = [&](int kb, int buf) {
        async_copy_b128(lds_off(&Ks[buf][kr][kc]),
                        (unsigned)(((kb + kr) * HD + kc) * 2), kb64);
        async_copy_b128(lds_off(&Ks[buf][kr][kc + 8]),
                        (unsigned)(((kb + kr) * HD + kc + 8) * 2), kb64);
        async_copy_b128(lds_off(&Vs[buf][vr][vc]),
                        (unsigned)((vr * S_ + kb + vc) * 2), vb64);
        async_copy_b128(lds_off(&Vs[buf][vr][vc + 8]),
                        (unsigned)((vr * S_ + kb + vc + 8) * 2), vb64);
    };

    // Q A-fragments held in registers for the whole pass
    v16h qa0 = load_afrag(Qg + (size_t)(qrow0 + n16) * HD, g);
    v16h qa1 = load_afrag(Qg + (size_t)(qrow0 + n16) * HD + 32, g);

    v8f ctx[4];
#pragma unroll
    for (int j = 0; j < 4; ++j) ctx[j] = (v8f)(0.0f);
    float rmax[8], rsum[8];
#pragma unroll
    for (int r = 0; r < 8; ++r) { rmax[r] = -3.0e38f; rsum[r] = 0.0f; }

    issue_kv(0, 0);
    for (int kb = 0, buf = 0; kb < S_; kb += 32, buf ^= 1) {
        wait_async0();
        __syncthreads();
        if (kb + 32 < S_) issue_kv(kb + 32, buf ^ 1);

        // scores for keys [kb,kb+16) and [kb+16,kb+32)
        v8f s0 = (v8f)(0.0f), s1 = (v8f)(0.0f);
        v16h bk00 = load_bfrag(&Ks[buf][n16][0], g);
        v16h bk01 = load_bfrag(&Ks[buf][n16][32], g);
        v16h bk10 = load_bfrag(&Ks[buf][16 + n16][0], g);
        v16h bk11 = load_bfrag(&Ks[buf][16 + n16][32], g);
        s0 = wmma_f16(qa0, bk00, s0); s0 = wmma_f16(qa1, bk01, s0);
        s1 = wmma_f16(qa0, bk10, s1); s1 = wmma_f16(qa1, bk11, s1);

        float mk0 = mb[kb + n16], mk1 = mb[kb + 16 + n16];
#pragma unroll
        for (int r = 0; r < 8; ++r) {          // row m = r + 8*g
            float a0 = s0[r] * 0.125f + mk0;   // 1/sqrt(64)
            float a1 = s1[r] * 0.125f + mk1;
            float tm = fmaxf(a0, a1);
            tm = fmaxf(tm, __shfl_xor(tm, 1));
            tm = fmaxf(tm, __shfl_xor(tm, 2));
            tm = fmaxf(tm, __shfl_xor(tm, 4));
            tm = fmaxf(tm, __shfl_xor(tm, 8));
            float nm = fmaxf(rmax[r], tm);
            float sc = __expf(rmax[r] - nm);
            rmax[r] = nm;
            float p0 = __expf(a0 - nm);
            float p1 = __expf(a1 - nm);
            float ps = p0 + p1;
            ps += __shfl_xor(ps, 1);
            ps += __shfl_xor(ps, 2);
            ps += __shfl_xor(ps, 4);
            ps += __shfl_xor(ps, 8);
            rsum[r] = rsum[r] * sc + ps;
#pragma unroll
            for (int j = 0; j < 4; ++j) ctx[j][r] *= sc;
            Pl[w][r + 8 * g][n16]      = (_Float16)p0;
            Pl[w][r + 8 * g][16 + n16] = (_Float16)p1;
        }
        // P store -> P load hazard within the wave (split DS counter)
        asm volatile("s_wait_dscnt 0x0" ::: "memory");

        v16h pa = load_afrag(&Pl[w][n16][0], g);
#pragma unroll
        for (int j = 0; j < 4; ++j) {
            v16h bv_ = load_bfrag(&Vs[buf][j * 16 + n16][0], g);
            ctx[j] = wmma_f16(pa, bv_, ctx[j]);
        }
        __syncthreads();        // done reading buf before it is refilled
    }

    // normalize and write f32 output [b][s][h*64+hd]
    float* ob = out + (size_t)b * S_ * H_;
#pragma unroll
    for (int j = 0; j < 4; ++j) {
        int col = h * HD + j * 16 + n16;
#pragma unroll
        for (int r = 0; r < 8; ++r) {
            int srow = qrow0 + r + 8 * g;
            ob[(size_t)srow * H_ + col] = ctx[j][r] / rsum[r];
        }
    }
}

// ---- host launch --------------------------------------------------------
extern "C" void kernel_launch(void* const* d_in, const int* in_sizes, int n_in,
                              void* d_out, int out_size, void* d_ws, size_t ws_size,
                              hipStream_t stream)
{
    const float* X    = (const float*)d_in[0];
    const float* mask = (const float*)d_in[1];
    const float* Wq   = (const float*)d_in[2];
    const float* bq   = (const float*)d_in[3];
    const float* Wk   = (const float*)d_in[4];
    const float* bk   = (const float*)d_in[5];
    const float* Wv   = (const float*)d_in[6];
    const float* bv   = (const float*)d_in[7];
    float* out = (float*)d_out;

    char* ws = (char*)d_ws;
    _Float16* Xh  = (_Float16*)(ws);                        // 16 MiB
    _Float16* Wtq = (_Float16*)(ws + (16ull << 20));        //  2 MiB each
    _Float16* Wtk = (_Float16*)(ws + (18ull << 20));
    _Float16* Wtv = (_Float16*)(ws + (20ull << 20));
    _Float16* Qb  = (_Float16*)(ws + (22ull << 20));        // 16 MiB each
    _Float16* Kb  = (_Float16*)(ws + (38ull << 20));
    _Float16* Vtb = (_Float16*)(ws + (54ull << 20));

    int n8 = (B_ * S_ * H_) / 8;
    cvt_f32_f16<<<n8 / 256, 256, 0, stream>>>(X, Xh, n8);

    transpose_w<<<dim3(32, 32, 3), dim3(32, 8), 0, stream>>>(Wq, Wk, Wv, Wtq, Wtk, Wtv);

    qkv_gemm<<<dim3(8, 64, 3), 256, 0, stream>>>(Xh, Wtq, Wtk, Wtv,
                                                 bq, bk, bv, Qb, Kb, Vtb);

    flash_attn<<<dim3(S_ / 64, NH, B_), 128, 0, stream>>>(Qb, Kb, Vtb, mask, out);
}
